// RelNetConvAttention_3504693314161
// MI455X (gfx1250) — matrix operand order repaired and checked by reading, not compiled, not run on hardware
//
#include <hip/hip_runtime.h>
#include <hip/hip_bf16.h>

// ---------------------------------------------------------------------------
// Relation Network (Sort-of-CLEVR) forward for MI455X (gfx1250).
// The pairwise g-MLP (~78 GFLOP of the ~80 GFLOP total) runs on WMMA
// v_wmma_f32_16x16x32_f16 with fp32 accumulation, fully fused in LDS.
// g-weights are pre-converted to f16 in fragment-swizzled order once, then
// each block streams them into double-buffered LDS with
// GLOBAL_LOAD_ASYNC_TO_LDS_B128, overlapped with the WMMA chunk loop.
// ---------------------------------------------------------------------------

typedef __attribute__((ext_vector_type(16))) _Float16 v16h;
typedef __attribute__((ext_vector_type(8)))  _Float16 v8h;
typedef __attribute__((ext_vector_type(2)))  _Float16 v2h;
typedef __attribute__((ext_vector_type(8)))  float    v8f;

#define B_   256
#define FS_  24
#define QD_  128
#define AV_  10

// ---------------------------------------------------------------- zero fill
__global__ void k_zero(float* __restrict__ p, int n) {
  int i = blockIdx.x * 256 + threadIdx.x;
  if (i < n) p[i] = 0.f;
}

// -------------------------------------------------- conv (stride2,pad1) + BN stats
template <int IC, int IH, int OH>
__global__ __launch_bounds__(256) void k_conv(
    const float* __restrict__ in, const float* __restrict__ w,
    const float* __restrict__ bias,
    const float* __restrict__ isc, const float* __restrict__ ish,
    float* __restrict__ out, float* __restrict__ bsum, float* __restrict__ bsq) {
  __shared__ float s1[256], s2[256];
  const int tid = threadIdx.x;
  const int oc = blockIdx.y, b = blockIdx.z;
  const int s = blockIdx.x * 256 + tid;
  const bool act = s < OH * OH;
  float v = 0.f;
  if (act) {
    const int oh = s / OH, ow = s % OH;
    float acc = bias[oc];
    for (int ic = 0; ic < IC; ++ic) {
      const float sc = isc ? isc[ic] : 1.f;
      const float sh = ish ? ish[ic] : 0.f;
      const float* ip = in + (size_t)(b * IC + ic) * IH * IH;
      const float* wp = w + (oc * IC + ic) * 9;
#pragma unroll
      for (int kh = 0; kh < 3; ++kh) {
        const int ih = oh * 2 - 1 + kh;
        if (ih < 0 || ih >= IH) continue;
#pragma unroll
        for (int kw = 0; kw < 3; ++kw) {
          const int iw = ow * 2 - 1 + kw;
          if (iw < 0 || iw >= IH) continue;
          acc += (ip[ih * IH + iw] * sc + sh) * wp[kh * 3 + kw];
        }
      }
    }
    v = fmaxf(acc, 0.f);
    out[(((size_t)b * FS_ + oc) * OH + (s / OH)) * OH + (s % OH)] = v;
  }
  s1[tid] = act ? v : 0.f;
  s2[tid] = act ? v * v : 0.f;
  __syncthreads();
  for (int st = 128; st > 0; st >>= 1) {
    if (tid < st) { s1[tid] += s1[tid + st]; s2[tid] += s2[tid + st]; }
    __syncthreads();
  }
  if (tid == 0) { atomicAdd(&bsum[oc], s1[0]); atomicAdd(&bsq[oc], s2[0]); }
}

// ------------------------------------------- BN finalize: per-channel scale/shift
__global__ void k_bnfin(const float* __restrict__ bsum, const float* __restrict__ bsq,
                        const float* __restrict__ g, const float* __restrict__ bta,
                        float* __restrict__ sc, float* __restrict__ sh, float invN) {
  const int t = threadIdx.x;
  if (t < FS_) {
    const float m = bsum[t] * invN;
    const float var = bsq[t] * invN - m * m;
    const float s = g[t] * rsqrtf(var + 1e-5f);
    sc[t] = s;
    sh[t] = bta[t] - m * s;
  }
}

// ------------------------------------------- dynamic per-sample conv weights
__global__ __launch_bounds__(256) void k_dynw(
    const float* __restrict__ qst, const float* __restrict__ qaww,
    const float* __restrict__ qawb, float* __restrict__ wdyn) {
  const int idx = blockIdx.x * 256 + threadIdx.x;
  if (idx >= B_ * 5184) return;
  const int b = idx / 5184, g = idx % 5184;
  float acc = qawb[g];
  for (int k = 0; k < QD_; ++k) acc += qst[k * B_ + b] * qaww[k * 5184 + g];
  wdyn[idx] = fmaxf(acc, 0.f);
}

// ------------------------------------------- grouped dynamic conv + BN4 stats
__global__ __launch_bounds__(32) void k_dynconv(
    const float* __restrict__ x3, const float* __restrict__ wdyn,
    const float* __restrict__ sc3, const float* __restrict__ sh3,
    float* __restrict__ x4, float* __restrict__ bsum, float* __restrict__ bsq) {
  __shared__ float s1[32], s2[32];
  const int t = threadIdx.x;
  const int oc = blockIdx.x, b = blockIdx.y;
  float v = 0.f;
  if (t < 25) {
    const int oh = t / 5, ow = t % 5;
    float acc = 0.f;
    const float* wp = wdyn + (size_t)b * 5184 + oc * 216;
    for (int ic = 0; ic < FS_; ++ic) {
      const float* ip = x3 + (size_t)(b * FS_ + ic) * 100;
      const float sc = sc3[ic], sh = sh3[ic];
#pragma unroll
      for (int kh = 0; kh < 3; ++kh) {
        const int ih = oh * 2 - 1 + kh;
        if (ih < 0 || ih >= 10) continue;
#pragma unroll
        for (int kw = 0; kw < 3; ++kw) {
          const int iw = ow * 2 - 1 + kw;
          if (iw < 0 || iw >= 10) continue;
          acc += (ip[ih * 10 + iw] * sc + sh) * wp[ic * 9 + kh * 3 + kw];
        }
      }
    }
    v = fmaxf(acc, 0.f);
    x4[((size_t)b * FS_ + oc) * 25 + t] = v;
  }
  s1[t] = (t < 25) ? v : 0.f;
  s2[t] = (t < 25) ? v * v : 0.f;
  __syncthreads();
  for (int st = 16; st > 0; st >>= 1) {
    if (t < st) { s1[t] += s1[t + st]; s2[t] += s2[t + st]; }
    __syncthreads();
  }
  if (t == 0) { atomicAdd(&bsum[oc], s1[0]); atomicAdd(&bsq[oc], s2[0]); }
}

// ------------------------------------------- qc = relu(q @ qcw + qcb)
__global__ __launch_bounds__(256) void k_qc(
    const float* __restrict__ qst, const float* __restrict__ qcw,
    const float* __restrict__ qcb, float* __restrict__ qc) {
  const int idx = blockIdx.x * 256 + threadIdx.x;
  if (idx >= B_ * QD_) return;
  const int b = idx >> 7, j = idx & 127;
  float acc = qcb[j];
  for (int k = 0; k < QD_; ++k) acc += qst[k * B_ + b] * qcw[k * QD_ + j];
  qc[idx] = fmaxf(acc, 0.f);
}

// ------------------------------------------- xf = [bn4(x4) | coords]  [B,25,26]
__global__ __launch_bounds__(256) void k_buildxf(
    const float* __restrict__ x4, const float* __restrict__ sc4,
    const float* __restrict__ sh4, float* __restrict__ xf) {
  const int idx = blockIdx.x * 256 + threadIdx.x;
  if (idx >= B_ * 25 * 26) return;
  const int b = idx / 650, rem = idx % 650, o = rem / 26, d = rem % 26;
  float v;
  if (d < FS_)      v = x4[((size_t)b * FS_ + d) * 25 + o] * sc4[d] + sh4[d];
  else if (d == 24) v = ((o / 5) - 2) * 0.5f;
  else              v = ((o % 5) - 2) * 0.5f;
  xf[idx] = v;
}

// ---------------------------------------------------------------------------
// One-time: convert a [Kreal x 256] f32 weight matrix into f16, fragment-
// swizzled per 32-K chunk so that per-lane B fragments are contiguous v16h:
//   out[(chunk*512 + ct*32 + lane)*16 + e] = W[chunk*32 + 16*hi + e][ct*16+lm]
// (zero-padded rows >= Kreal).
// ---------------------------------------------------------------------------
__global__ __launch_bounds__(256) void k_prepw(
    const float* __restrict__ W, _Float16* __restrict__ out,
    int Kreal, int nchunks) {
  const int idx = blockIdx.x * 256 + threadIdx.x;
  if (idx >= nchunks * 8192) return;
  const int chunk = idx >> 13;
  const int rem = idx & 8191;
  const int slot = rem >> 4, e = rem & 15;
  const int ct = slot >> 5, lane = slot & 31;
  const int hi = lane >> 4, lm = lane & 15;
  const int k = (chunk << 5) + (hi << 4) + e;
  const int n = (ct << 4) + lm;
  out[idx] = (_Float16)((k < Kreal) ? W[k * 256 + n] : 0.f);
}

// ---------------------------------------------------------------------------
// Async contiguous 16 KB copy: global (pre-swizzled f16 weights) -> LDS.
// Each of 256 threads issues 4 x GLOBAL_LOAD_ASYNC_TO_LDS_B128 (ASYNCcnt).
// LDS address operand = low 32 bits of the flat shared address.
// ---------------------------------------------------------------------------
__device__ __forceinline__ void async_copy_chunk(
    const _Float16* __restrict__ src, _Float16* dst_lds, int tid) {
  const unsigned lbase = (unsigned)(size_t)dst_lds;
  const char* g = (const char*)src;
#pragma unroll
  for (int u = 0; u < 4; ++u) {
    const unsigned off = (unsigned)(tid + u * 256) * 16u;
    asm volatile("global_load_async_to_lds_b128 %0, %1, off"
                 :: "v"(lbase + off), "v"(g + off)
                 : "memory");
  }
}

// ---------------------------------------------------------------------------
// WMMA g-MLP. One block = (sample b, chunk of 64 pair-rows), 8 waves.
// Wave w: row-pair rp = w>>2 (2 x 16-row tiles), col-quarter cq = w&3
// (4 x 16-col tiles) -> 8 accumulators.
// Fragment packing (A and B share one contiguous K order, not AMD's table order):
//   A (16x32):  lane: M = lane&15, hi = lane>>4; elem e: K = 16*hi + e
//               (= two contiguous 8-half runs -> 2 x ds_load_b128)
//   B (32x16):  lane: N = lane&15, hi = lane>>4; elem e: K = 16*hi + e
//               (pre-swizzled weights -> contiguous v16h in LDS)
//   C/D(16x16): lane: N = lane&15; reg r: M = r + 8*hi
// ---------------------------------------------------------------------------
__device__ __forceinline__ void g_layer(
    const _Float16* __restrict__ Wsw, const float* __restrict__ bias,
    _Float16* s_act, v16h* s_wkv /* 2 x 512 frags (double buffer) */,
    int nchunks, int tid, bool last, float* __restrict__ xg_row, int rowbase) {
  const int wave = tid >> 5, lane = tid & 31;
  const int rp = wave >> 2, cq = wave & 3;
  const int hi = lane >> 4, lm = lane & 15;

  v8f acc[2][4];
#pragma unroll
  for (int i = 0; i < 2; ++i)
#pragma unroll
    for (int j = 0; j < 4; ++j)
#pragma unroll
      for (int r = 0; r < 8; ++r) acc[i][j][r] = 0.f;

  // prologue: async-stage chunk 0
  async_copy_chunk(Wsw, (_Float16*)s_wkv, tid);
  asm volatile("s_wait_asynccnt 0" ::: "memory");
  __syncthreads();

  for (int kc = 0; kc < nchunks; ++kc) {
    const int kbase = kc << 5;
    const v16h* cur = s_wkv + (kc & 1) * 512;
    // overlap: async-stage next chunk into the other buffer
    if (kc + 1 < nchunks)
      async_copy_chunk(Wsw + (size_t)(kc + 1) * 8192,
                       (_Float16*)(s_wkv + ((kc + 1) & 1) * 512), tid);

    // A fragments: two contiguous 8-half runs per lane -> wide LDS loads
    v16h a[2];
#pragma unroll
    for (int i = 0; i < 2; ++i) {
      const int arow = ((rp << 1) + i) * 16 + lm;
      const v8h* pa = (const v8h*)(s_act + arow * 256 + kbase);
      const v8h alo = pa[2 * hi];
      const v8h ahi = pa[2 * hi + 1];
      a[i] = __builtin_shufflevector(alo, ahi, 0, 1, 2, 3, 4, 5, 6, 7,
                                     8, 9, 10, 11, 12, 13, 14, 15);
    }
    // B fragments: fragment-contiguous v16h reads
    v16h bf[4];
#pragma unroll
    for (int j = 0; j < 4; ++j) bf[j] = cur[((cq << 2) + j) * 32 + lane];

#pragma unroll
    for (int i = 0; i < 2; ++i)
#pragma unroll
      for (int j = 0; j < 4; ++j)
        acc[i][j] = __builtin_amdgcn_wmma_f32_16x16x32_f16(
            false, a[i], false, bf[j], (short)0, acc[i][j], false, false);

    // wait for our async stores, then one barrier per chunk
    asm volatile("s_wait_asynccnt 0" ::: "memory");
    __syncthreads();
  }

#pragma unroll
  for (int i = 0; i < 2; ++i)
#pragma unroll
    for (int j = 0; j < 4; ++j) {
      const int n = ((cq << 2) + j) * 16 + lm;
      const float bv = bias[n];
      if (!last) {
#pragma unroll
        for (int r = 0; r < 8; ++r) {
          const float v = fmaxf(acc[i][j][r] + bv, 0.f);
          const int row = ((rp << 1) + i) * 16 + r + (hi << 3);
          s_act[row * 256 + n] = (_Float16)v;
        }
      } else {
        float s = 0.f;
#pragma unroll
        for (int r = 0; r < 8; ++r) {
          const int row = ((rp << 1) + i) * 16 + r + (hi << 3);
          if (rowbase + row < 625) s += fmaxf(acc[i][j][r] + bv, 0.f);
        }
        atomicAdd(&xg_row[n], s);
      }
    }
  __syncthreads();
}

__global__ __launch_bounds__(256) void k_gmlp(
    const float* __restrict__ xf, const float* __restrict__ qc,
    const _Float16* __restrict__ w1, const float* __restrict__ g1b,
    const _Float16* __restrict__ w2, const float* __restrict__ g2b,
    const _Float16* __restrict__ w3, const float* __restrict__ g3b,
    const _Float16* __restrict__ w4, const float* __restrict__ g4b,
    float* __restrict__ xg) {
  __shared__ v8h  s_actv[64 * 256 / 8];  // 32 KB activations (in-place per layer)
  __shared__ v16h s_wkv[2 * 512];        // 2 x 16 KB double-buffered weight chunk
  _Float16* s_act = (_Float16*)s_actv;
  const int b = blockIdx.x;
  const int rowbase = blockIdx.y * 64;
  const int tid = threadIdx.x;

  // Build layer-1 input: row = pair (p,q); feat = [xf[b,q] | xf[b,p] | qc[b] | 0pad]
  for (int i = tid; i < 64 * 96; i += 256) {
    const int row = i / 96, dp = i % 96;
    const int pr = rowbase + row;
    const int d0 = dp << 1;
    v2h pk;
#pragma unroll
    for (int u = 0; u < 2; ++u) {
      const int d = d0 + u;
      float v = 0.f;
      if (pr < 625 && d < 180) {
        const int p = pr / 25, q = pr % 25;
        if (d < 26)       v = xf[(b * 25 + q) * 26 + d];
        else if (d < 52)  v = xf[(b * 25 + p) * 26 + (d - 26)];
        else              v = qc[b * QD_ + (d - 52)];
      }
      pk[u] = (_Float16)v;
    }
    *(v2h*)(s_act + row * 256 + d0) = pk;
  }
  // (s_act consumption is guarded by the prologue barrier inside g_layer)
  g_layer(w1, g1b, s_act, s_wkv, 6, tid, false, nullptr, 0);
  g_layer(w2, g2b, s_act, s_wkv, 8, tid, false, nullptr, 0);
  g_layer(w3, g3b, s_act, s_wkv, 8, tid, false, nullptr, 0);
  g_layer(w4, g4b, s_act, s_wkv, 8, tid, true, xg + (size_t)b * 256, rowbase);
}

// ------------------------------------------- f-MLP + log_softmax (per sample)
__global__ __launch_bounds__(256) void k_fmlp(
    const float* __restrict__ xg,
    const float* __restrict__ f1w, const float* __restrict__ f1b,
    const float* __restrict__ o2w, const float* __restrict__ o2b,
    const float* __restrict__ o3w, const float* __restrict__ o3b,
    float* __restrict__ out) {
  __shared__ float sx[256], sy[256], lg[AV_], red[2];
  const int t = threadIdx.x, b = blockIdx.x;
  sx[t] = xg[(size_t)b * 256 + t];
  __syncthreads();
  float acc = f1b[t];
  for (int k = 0; k < 256; ++k) acc += sx[k] * f1w[k * 256 + t];
  sy[t] = fmaxf(acc, 0.f);
  __syncthreads();
  sx[t] = sy[t];
  __syncthreads();
  acc = o2b[t];
  for (int k = 0; k < 256; ++k) acc += sx[k] * o2w[k * 256 + t];
  sy[t] = fmaxf(acc, 0.f);
  __syncthreads();
  sx[t] = sy[t];
  __syncthreads();
  if (t < AV_) {
    float a2 = o3b[t];
    for (int k = 0; k < 256; ++k) a2 += sx[k] * o3w[k * AV_ + t];
    lg[t] = a2;
  }
  __syncthreads();
  if (t == 0) {
    float m = lg[0];
    for (int i = 1; i < AV_; ++i) m = fmaxf(m, lg[i]);
    float s = 0.f;
    for (int i = 0; i < AV_; ++i) s += expf(lg[i] - m);
    red[0] = m;
    red[1] = logf(s);
  }
  __syncthreads();
  if (t < AV_) out[b * AV_ + t] = lg[t] - red[0] - red[1];
}

// ---------------------------------------------------------------------------
extern "C" void kernel_launch(void* const* d_in, const int* in_sizes, int n_in,
                              void* d_out, int out_size, void* d_ws, size_t ws_size,
                              hipStream_t stream) {
  const float* img  = (const float*)d_in[0];
  const float* qst  = (const float*)d_in[1];
  const float* c1w  = (const float*)d_in[2];
  const float* c1b  = (const float*)d_in[3];
  const float* bn1g = (const float*)d_in[4];
  const float* bn1b = (const float*)d_in[5];
  const float* c2w  = (const float*)d_in[6];
  const float* c2b  = (const float*)d_in[7];
  const float* bn2g = (const float*)d_in[8];
  const float* bn2b = (const float*)d_in[9];
  const float* c3w  = (const float*)d_in[10];
  const float* c3b  = (const float*)d_in[11];
  const float* bn3g = (const float*)d_in[12];
  const float* bn3b = (const float*)d_in[13];
  const float* bn4g = (const float*)d_in[14];
  const float* bn4b = (const float*)d_in[15];
  const float* qaww = (const float*)d_in[16];
  const float* qawb = (const float*)d_in[17];
  const float* qcw  = (const float*)d_in[18];
  const float* qcb  = (const float*)d_in[19];
  const float* g1w  = (const float*)d_in[20];
  const float* g1b  = (const float*)d_in[21];
  const float* g2w  = (const float*)d_in[22];
  const float* g2b  = (const float*)d_in[23];
  const float* g3w  = (const float*)d_in[24];
  const float* g3b  = (const float*)d_in[25];
  const float* g4w  = (const float*)d_in[26];
  const float* g4b  = (const float*)d_in[27];
  const float* f1w  = (const float*)d_in[28];
  const float* f1b  = (const float*)d_in[29];
  const float* o2w  = (const float*)d_in[30];
  const float* o2b  = (const float*)d_in[31];
  const float* o3w  = (const float*)d_in[32];
  const float* o3b  = (const float*)d_in[33];
  float* out = (float*)d_out;

  // workspace layout (floats; every region size divisible by 4 -> 16B aligned)
  constexpr size_t N_X1 = 256ull * 24 * 38 * 38;
  constexpr size_t N_X2 = 256ull * 24 * 19 * 19;
  constexpr size_t N_X3 = 256ull * 24 * 10 * 10;
  constexpr size_t N_WD = 256ull * 5184;
  constexpr size_t N_X4 = 256ull * 24 * 25;
  constexpr size_t N_QC = 256ull * 128;
  constexpr size_t N_XF = 256ull * 25 * 26;
  constexpr size_t N_XG = 256ull * 256;
  float* ws = (float*)d_ws;
  float* x1   = ws;
  float* x2   = x1 + N_X1;
  float* x3   = x2 + N_X2;
  float* wdyn = x3 + N_X3;
  float* x4   = wdyn + N_WD;
  float* qc   = x4 + N_X4;
  float* xf   = qc + N_QC;
  float* xg   = xf + N_XF;
  float* st   = xg + N_XG;  // 4 BNs x (sum[24], sq[24], scale[24], shift[24])
  float* sum1 = st + 0 * 96;   float* sq1 = sum1 + 24; float* sc1 = sum1 + 48; float* sh1 = sum1 + 72;
  float* sum2 = st + 1 * 96;   float* sq2 = sum2 + 24; float* sc2 = sum2 + 48; float* sh2 = sum2 + 72;
  float* sum3 = st + 2 * 96;   float* sq3 = sum3 + 24; float* sc3 = sum3 + 48; float* sh3 = sum3 + 72;
  float* sum4 = st + 3 * 96;   float* sq4 = sum4 + 24; float* sc4 = sum4 + 48; float* sh4 = sum4 + 72;
  // pre-swizzled f16 g-weights: L1 = 6 chunks, L2-4 = 8 chunks (8192 halves ea.)
  _Float16* wsw1 = (_Float16*)(st + 4 * 96);
  _Float16* wsw2 = wsw1 + 6 * 8192;
  _Float16* wsw3 = wsw2 + 8 * 8192;
  _Float16* wsw4 = wsw3 + 8 * 8192;

  // zero xg + BN stats (contiguous region)
  const int nz = (int)(N_XG + 4 * 96);
  k_zero<<<(nz + 255) / 256, 256, 0, stream>>>(xg, nz);

  // one-time f16 fragment-swizzle of the g-MLP weights
  k_prepw<<<192, 256, 0, stream>>>(g1w, wsw1, 180, 6);
  k_prepw<<<256, 256, 0, stream>>>(g2w, wsw2, 256, 8);
  k_prepw<<<256, 256, 0, stream>>>(g3w, wsw3, 256, 8);
  k_prepw<<<256, 256, 0, stream>>>(g4w, wsw4, 256, 8);

  // conv stack with training-mode BN (affine folded into consumer loads)
  k_conv<3, 75, 38><<<dim3(6, 24, 256), 256, 0, stream>>>(
      img, c1w, c1b, nullptr, nullptr, x1, sum1, sq1);
  k_bnfin<<<1, 32, 0, stream>>>(sum1, sq1, bn1g, bn1b, sc1, sh1, 1.f / (256.f * 38 * 38));
  k_conv<24, 38, 19><<<dim3(2, 24, 256), 256, 0, stream>>>(
      x1, c2w, c2b, sc1, sh1, x2, sum2, sq2);
  k_bnfin<<<1, 32, 0, stream>>>(sum2, sq2, bn2g, bn2b, sc2, sh2, 1.f / (256.f * 19 * 19));
  k_conv<24, 19, 10><<<dim3(1, 24, 256), 256, 0, stream>>>(
      x2, c3w, c3b, sc2, sh2, x3, sum3, sq3);
  k_bnfin<<<1, 32, 0, stream>>>(sum3, sq3, bn3g, bn3b, sc3, sh3, 1.f / (256.f * 10 * 10));

  // question-conditioned dynamic conv
  k_dynw<<<(256 * 5184 + 255) / 256, 256, 0, stream>>>(qst, qaww, qawb, wdyn);
  k_dynconv<<<dim3(24, 256), 32, 0, stream>>>(x3, wdyn, sc3, sh3, x4, sum4, sq4);
  k_bnfin<<<1, 32, 0, stream>>>(sum4, sq4, bn4g, bn4b, sc4, sh4, 1.f / (256.f * 25));

  // question embedding + object features
  k_qc<<<(256 * 128 + 255) / 256, 256, 0, stream>>>(qst, qcw, qcb, qc);
  k_buildxf<<<(256 * 25 * 26 + 255) / 256, 256, 0, stream>>>(x4, sc4, sh4, xf);

  // fused pairwise g-MLP (WMMA + async-to-LDS weight streaming)
  k_gmlp<<<dim3(256, 10), 256, 0, stream>>>(
      xf, qc, wsw1, g1b, wsw2, g2b, wsw3, g3b, wsw4, g4b, xg);

  // f-MLP + log_softmax
  k_fmlp<<<256, 256, 0, stream>>>(xg, f1w, f1b, o2w, o2b, o3w, o3b, out);
}
